// LSTMAutocomplete_60163901882394
// MI455X (gfx1250) — matrix operand order repaired and checked by reading, not compiled, not run on hardware
//
#include <hip/hip_runtime.h>
#include <hip/hip_bf16.h>
#include <stdint.h>
#include <stddef.h>

typedef __attribute__((ext_vector_type(16))) _Float16 v16h;
typedef __attribute__((ext_vector_type(8)))  _Float16 v8h;
typedef __attribute__((ext_vector_type(8)))  float    v8f;

#define VSZ   50000
#define EDIM  128
#define HDIM  128
#define BSZ   16
#define TSZ   256
#define MROWS (BSZ*TSZ)     /* 4096 rows, m = b*T + t */
#define GDIM  (4*HDIM)      /* 512 gate columns      */
#define NTV   (VSZ/16)      /* 3125 vocab N-tiles    */

// ---------------------------------------------------------------------------
// WMMA fragment loaders per CDNA5 ISA 7.12.2 (16-bit A 16x32, B 32x16).
// A (MxK, row-major, ld=K):  lane&15 = M row; lane>>4 selects K-half:
//   halves 0..7  = K[kb + half*8 .. +7], halves 8..15 = K[kb+16+half*8 .. +7]
// B (KxN) from row-major weight W[N][K]: lane&15 = N col; lane>>4 selects K-half:
//   halves 0..7  = K[kb + half*8 .. +7], halves 8..15 = K[kb+16+half*8 .. +7]
// ---------------------------------------------------------------------------
__device__ __forceinline__ v16h load_a_frag(const _Float16* A, int row, int ldk,
                                            int kblock, int lane) {
    const int half = lane >> 4;
    const _Float16* p = A + row * ldk + kblock;
    v8h lo = *(const v8h*)(p + half * 8);
    v8h hi = *(const v8h*)(p + 16 + half * 8);
    return __builtin_shufflevector(lo, hi, 0,1,2,3,4,5,6,7,8,9,10,11,12,13,14,15);
}

__device__ __forceinline__ v16h load_b_frag(const _Float16* W, int col, int ldk,
                                            int kblock, int lane) {
    const int kbase = kblock + ((lane >> 4) * 8);
    const _Float16* p = W + col * ldk + kbase;
    v8h lo = *(const v8h*)(p);
    v8h hi = *(const v8h*)(p + 16);
    return __builtin_shufflevector(lo, hi, 0,1,2,3,4,5,6,7,8,9,10,11,12,13,14,15);
}

// ---------------------------------------------------------------------------
// Elementwise f32 -> f16 conversion (weights / fc_w), grid-stride.
// ---------------------------------------------------------------------------
__global__ void f32_to_f16_kernel(const float* __restrict__ in,
                                  _Float16* __restrict__ out, int n) {
    int i = blockIdx.x * blockDim.x + threadIdx.x;
    int stride = gridDim.x * blockDim.x;
    for (; i < n; i += stride) out[i] = (_Float16)in[i];
}

// Embedding gather, f32 table -> f16 activation rows (m = b*T + t).
__global__ void embed_kernel(const int* __restrict__ x,
                             const float* __restrict__ emb,
                             _Float16* __restrict__ e16, int n) {
    int i = blockIdx.x * blockDim.x + threadIdx.x;
    int stride = gridDim.x * blockDim.x;
    for (; i < n; i += stride) {
        int m = i >> 7, e = i & 127;
        e16[i] = (_Float16)emb[x[m] * EDIM + e];
    }
}

// ---------------------------------------------------------------------------
// xp = A[4096,128] @ W[512,128]^T + (b_ih + b_hh), f32 out [4096][512].
// One wave per 16x16 tile; 8 waves / block; 8192 tiles -> 1024 blocks.
// ---------------------------------------------------------------------------
__global__ __launch_bounds__(256) void xp_gemm_kernel(
    const _Float16* __restrict__ A,   // [MROWS][128]
    const _Float16* __restrict__ W,   // [GDIM][128] row-major (N-major)
    const float* __restrict__ b1, const float* __restrict__ b2,
    float* __restrict__ xp)           // [MROWS][GDIM]
{
    const int lane = threadIdx.x & 31;
    const int w    = blockIdx.x * 8 + (threadIdx.x >> 5);
    const int mt   = w >> 5;                  // 0..255
    const int nt   = w & 31;                  // 0..31
    const int col  = nt * 16 + (lane & 15);
    const int arow = mt * 16 + (lane & 15);
    const int moff = (lane >= 16) ? 8 : 0;

    v8f acc = {};
#pragma unroll
    for (int q = 0; q < 4; ++q) {
        v16h a = load_a_frag(A, arow, 128, q * 32, lane);
        v16h b = load_b_frag(W, col, 128, q * 32, lane);
        acc = __builtin_amdgcn_wmma_f32_16x16x32_f16(false, a, false, b,
                                                     (short)0, acc, false, false);
    }
    const float bias = b1[col] + b2[col];
#pragma unroll
    for (int r = 0; r < 8; ++r) {
        int m = mt * 16 + r + moff;
        xp[m * GDIM + col] = acc[r] + bias;
    }
}

// ---------------------------------------------------------------------------
// Sequential LSTM layer: one persistent workgroup of 32 waves (1024 threads).
// Wave nt owns gate N-tile nt; W_hh f16 B-fragments live in registers for all
// 256 steps; h lives in LDS as f16 (shared A fragments); raw gates staged in
// LDS f32; cell state c lives in registers (2 elements / thread).
// ---------------------------------------------------------------------------
__global__ __launch_bounds__(1024) void lstm_seq_kernel(
    const float* __restrict__ xp,      // [MROWS][GDIM], row m = b*T + t
    const _Float16* __restrict__ Whh,  // [GDIM][128] f16
    _Float16* __restrict__ y16,        // [MROWS][HDIM] f16 outputs
    float* __restrict__ h_out,         // [2048] = [B][H] final h
    float* __restrict__ c_out)         // [2048] final c
{
    __shared__ _Float16 h16[BSZ * HDIM];   // 4 KB
    __shared__ float    gates[BSZ * GDIM]; // 32 KB

    const int tid  = threadIdx.x;
    const int lane = tid & 31;
    const int nt   = tid >> 5;                // 0..31
    const int col  = nt * 16 + (lane & 15);
    const int moff = (lane >= 16) ? 8 : 0;
    const int arow = lane & 15;               // batch row for A fragment

    v16h bfr[4];
#pragma unroll
    for (int q = 0; q < 4; ++q) bfr[q] = load_b_frag(Whh, col, 128, q * 32, lane);

    // h = 0, c = 0
    h16[tid]        = (_Float16)0.f;
    h16[tid + 1024] = (_Float16)0.f;
    float creg0 = 0.f, creg1 = 0.f;
    __syncthreads();

    for (int t = 0; t < TSZ; ++t) {
        // C := precomputed input projection for this timestep
        v8f acc;
#pragma unroll
        for (int r = 0; r < 8; ++r) {
            int b = r + moff;
            acc[r] = xp[(b * TSZ + t) * GDIM + col];
        }
        // gates += h @ W_hh^T
#pragma unroll
        for (int q = 0; q < 4; ++q) {
            v16h a = load_a_frag(h16, arow, HDIM, q * 32, lane);
            acc = __builtin_amdgcn_wmma_f32_16x16x32_f16(false, a, false, bfr[q],
                                                         (short)0, acc, false, false);
        }
#pragma unroll
        for (int r = 0; r < 8; ++r) gates[(r + moff) * GDIM + col] = acc[r];
        __syncthreads();

        // elementwise combine: 2048 (b,h) cells over 1024 threads
#pragma unroll
        for (int s = 0; s < 2; ++s) {
            int k = tid + s * 1024;
            int b = k >> 7, j = k & 127;
            const float* gb = &gates[b * GDIM];
            float ig = 1.f / (1.f + __expf(-gb[j]));
            float fg = 1.f / (1.f + __expf(-gb[HDIM + j]));
            float gg = tanhf(gb[2 * HDIM + j]);
            float og = 1.f / (1.f + __expf(-gb[3 * HDIM + j]));
            float cr = s ? creg1 : creg0;
            cr = fg * cr + ig * gg;
            float h = og * tanhf(cr);
            if (s) creg1 = cr; else creg0 = cr;
            h16[b * HDIM + j] = (_Float16)h;
            y16[(b * TSZ + t) * HDIM + j] = (_Float16)h;
            if (t == TSZ - 1) { h_out[k] = h; c_out[k] = cr; }
        }
        __syncthreads();
    }
}

// ---------------------------------------------------------------------------
// logits = Y[4096,128] @ fc_w[50000,128]^T + fc_b  (f32 out, 819 MB: the
// store stream is the roofline; fc_w16 (12.8 MB) is L2-resident).
// Each wave: 1 vocab N-tile (B-frags pinned in regs) x 8 M-tiles.
// 3125 * 32 = 100000 waves -> 12500 blocks of 8 waves.
// ---------------------------------------------------------------------------
__global__ __launch_bounds__(256) void logits_gemm_kernel(
    const _Float16* __restrict__ Y,    // [MROWS][128]
    const _Float16* __restrict__ Wf,   // [VSZ][128]
    const float* __restrict__ fcb,     // [VSZ]
    float* __restrict__ out)           // [MROWS][VSZ]
{
    const int lane = threadIdx.x & 31;
    const int w    = blockIdx.x * 8 + (threadIdx.x >> 5);
    if (w >= NTV * 32) return;
    const int nt   = w % NTV;                 // 0..3124
    const int mg   = w / NTV;                 // 0..31
    const int col  = nt * 16 + (lane & 15);
    const int moff = (lane >= 16) ? 8 : 0;

    v16h bfr[4];
#pragma unroll
    for (int q = 0; q < 4; ++q) bfr[q] = load_b_frag(Wf, col, 128, q * 32, lane);
    const float bias = fcb[col];

    for (int mi = 0; mi < 8; ++mi) {
        const int mt   = mg * 8 + mi;
        const int arow = mt * 16 + (lane & 15);
        v8f acc = {};
#pragma unroll
        for (int q = 0; q < 4; ++q) {
            v16h a = load_a_frag(Y, arow, 128, q * 32, lane);
            acc = __builtin_amdgcn_wmma_f32_16x16x32_f16(false, a, false, bfr[q],
                                                         (short)0, acc, false, false);
        }
#pragma unroll
        for (int r = 0; r < 8; ++r) {
            size_t m = (size_t)(mt * 16 + r + moff);
            out[m * VSZ + col] = acc[r] + bias;
        }
    }
}

// ---------------------------------------------------------------------------
// Orchestration
// ---------------------------------------------------------------------------
extern "C" void kernel_launch(void* const* d_in, const int* in_sizes, int n_in,
                              void* d_out, int out_size, void* d_ws, size_t ws_size,
                              hipStream_t stream) {
    (void)in_sizes; (void)n_in; (void)out_size; (void)ws_size;
    const int*   x    = (const int*)  d_in[0];
    const float* emb  = (const float*)d_in[1];
    const float* Wih0 = (const float*)d_in[2];
    const float* Whh0 = (const float*)d_in[3];
    const float* bih0 = (const float*)d_in[4];
    const float* bhh0 = (const float*)d_in[5];
    const float* Wih1 = (const float*)d_in[6];
    const float* Whh1 = (const float*)d_in[7];
    const float* bih1 = (const float*)d_in[8];
    const float* bhh1 = (const float*)d_in[9];
    const float* fcw  = (const float*)d_in[10];
    const float* fcb  = (const float*)d_in[11];
    float* out = (float*)d_out;

    // workspace carve-up (all chunks 16B-aligned sizes)
    char* ws = (char*)d_ws;
    _Float16* e16   = (_Float16*)ws; ws += (size_t)MROWS * EDIM * 2;     // 1   MB
    _Float16* wih0h = (_Float16*)ws; ws += (size_t)GDIM * EDIM * 2;      // 128 KB
    _Float16* whh0h = (_Float16*)ws; ws += (size_t)GDIM * HDIM * 2;
    _Float16* wih1h = (_Float16*)ws; ws += (size_t)GDIM * HDIM * 2;
    _Float16* whh1h = (_Float16*)ws; ws += (size_t)GDIM * HDIM * 2;
    _Float16* fcwh  = (_Float16*)ws; ws += (size_t)VSZ * HDIM * 2;       // 12.8 MB
    _Float16* y0h   = (_Float16*)ws; ws += (size_t)MROWS * HDIM * 2;     // 1 MB
    _Float16* y1h   = (_Float16*)ws; ws += (size_t)MROWS * HDIM * 2;     // 1 MB
    float*    xp0   = (float*)ws;    ws += (size_t)MROWS * GDIM * 4;     // 8 MB
    float*    xp1   = (float*)ws;    ws += (size_t)MROWS * GDIM * 4;     // 8 MB

    float* h_out = out + (size_t)MROWS * VSZ;        // [2][16][128]
    float* c_out = h_out + 2 * BSZ * HDIM;           // [2][16][128]

    // 1) weight conversions to f16
    f32_to_f16_kernel<<<256, 256, 0, stream>>>(Wih0, wih0h, GDIM * EDIM);
    f32_to_f16_kernel<<<256, 256, 0, stream>>>(Whh0, whh0h, GDIM * HDIM);
    f32_to_f16_kernel<<<256, 256, 0, stream>>>(Wih1, wih1h, GDIM * HDIM);
    f32_to_f16_kernel<<<256, 256, 0, stream>>>(Whh1, whh1h, GDIM * HDIM);
    f32_to_f16_kernel<<<4096, 256, 0, stream>>>(fcw, fcwh, VSZ * HDIM);

    // 2) embedding gather -> f16
    embed_kernel<<<2048, 256, 0, stream>>>(x, emb, e16, MROWS * EDIM);

    // 3) layer 0: input projection, then sequential recurrence
    xp_gemm_kernel<<<1024, 256, 0, stream>>>(e16, wih0h, bih0, bhh0, xp0);
    lstm_seq_kernel<<<1, 1024, 0, stream>>>(xp0, whh0h, y0h, h_out, c_out);

    // 4) layer 1
    xp_gemm_kernel<<<1024, 256, 0, stream>>>(y0h, wih1h, bih1, bhh1, xp1);
    lstm_seq_kernel<<<1, 1024, 0, stream>>>(xp1, whh1h, y1h,
                                            h_out + BSZ * HDIM, c_out + BSZ * HDIM);

    // 5) vocab projection (store-bandwidth bound)
    logits_gemm_kernel<<<12500, 256, 0, stream>>>(y1h, fcwh, fcb, out);
}
